// DynaRepViTBlock_79697413145228
// MI455X (gfx1250) — matrix operand order repaired and checked by reading, not compiled, not run on hardware
//
#include <hip/hip_runtime.h>
#include <hip/hip_bf16.h>
#include <math.h>

// ---------------- constants ----------------
#define B_    64
#define C_    384
#define HW_   1024          // 32*32
#define HID_  768
#define MID_  24
#define M_    (B_ * HW_)    // 65536
#define EPS_  1e-5f

typedef __attribute__((ext_vector_type(16))) __bf16 v16bf;
typedef __attribute__((ext_vector_type(8)))  __bf16 bf16x8;
typedef __attribute__((ext_vector_type(8)))  float  v8f;

__device__ __forceinline__ __bf16 f2bf(float f) {
    unsigned u = __builtin_bit_cast(unsigned, f);
    unsigned r = (u + 0x7FFFu + ((u >> 16) & 1u)) >> 16;
    return __builtin_bit_cast(__bf16, (unsigned short)r);
}
__device__ __forceinline__ float gelu_exact(float x) {
    return 0.5f * x * (1.0f + erff(x * 0.70710678118654752f));
}
__device__ __forceinline__ float sigmoidf_(float x) {
    return 1.0f / (1.0f + expf(-x));
}

// CDNA5 async global->LDS copy (16 bytes per lane), tracked by ASYNCcnt.
// dst-first operand order per gfx1250 disasm convention; VDST = LDS address VGPR.
__device__ __forceinline__ void async_copy_b128(unsigned lds_addr, const void* gptr) {
    unsigned long long ga = (unsigned long long)(size_t)gptr;
    asm volatile("global_load_async_to_lds_b128 %0, %1, off"
                 :: "v"(lds_addr), "v"(ga) : "memory");
}
__device__ __forceinline__ void wait_asynccnt0() {
    asm volatile("s_wait_asynccnt 0x0" ::: "memory");
}

// ---------------- utility kernels ----------------
__global__ void zero_kernel(float* a, int n) {
    int i = blockIdx.x * blockDim.x + threadIdx.x;
    if (i < n) a[i] = 0.0f;
}

// mean over H,W per (b,c): one block per (b,c)
__global__ void pool_kernel(const float* __restrict__ x, float* __restrict__ p) {
    __shared__ float red[256];
    int bc = blockIdx.x;
    const float* xp = x + (size_t)bc * HW_;
    float s = 0.f;
    for (int k = threadIdx.x; k < HW_; k += 256) s += xp[k];
    red[threadIdx.x] = s;
    __syncthreads();
    for (int off = 128; off; off >>= 1) {
        if (threadIdx.x < off) red[threadIdx.x] += red[threadIdx.x + off];
        __syncthreads();
    }
    if (threadIdx.x == 0) p[bc] = red[0] * (1.0f / HW_);
}

// finalize BN stats: mean = sum/cnt, rstd = rsqrt(var+eps)
__global__ void finalize_kernel(const float* sum, const float* ss, float invCnt,
                                int len, float* mean, float* rstd) {
    int i = blockIdx.x * blockDim.x + threadIdx.x;
    if (i < len) {
        float m = sum[i] * invCnt;
        float v = ss[i] * invCnt - m * m;
        mean[i] = m;
        rstd[i] = rsqrtf(v + EPS_);
    }
}

__global__ void pack_bf16_kernel(const float* __restrict__ src, __bf16* __restrict__ dst, int n) {
    for (int i = blockIdx.x * blockDim.x + threadIdx.x; i < n; i += gridDim.x * blockDim.x)
        dst[i] = f2bf(src[i]);
}

// ---------------- controller (dc + kp MLPs), one block ----------------
__global__ void controller_kernel(
    const float* __restrict__ p,
    const float* dc_w1, const float* dc_g1, const float* dc_b1,
    const float* dc_w2, const float* dc_g2, const float* dc_b2,
    const float* kp_w1, const float* kp_g1, const float* kp_b1,
    const float* kp_w2, const float* kp_g2, const float* kp_b2,
    float* gate, float* kw0, float* kw1)
{
    __shared__ float sh_h[B_ * 8];
    __shared__ float sh_a[B_ * 4];
    __shared__ float sh_s[B_];
    __shared__ float sh_l[B_ * 2];
    __shared__ float sc[8], sf[8];
    int tid = threadIdx.x;

    // dc layer 1: [B,8]
    for (int t = tid; t < B_ * 8; t += 256) {
        int b = t >> 3, j = t & 7;
        float s = 0.f;
        const float* wr = dc_w1 + j * C_;
        const float* pr = p + b * C_;
        for (int c = 0; c < C_; ++c) s += wr[c] * pr[c];
        sh_h[t] = s;
    }
    __syncthreads();
    if (tid < 8) {
        float m = 0.f, q = 0.f;
        for (int b = 0; b < B_; ++b) { float v = sh_h[b * 8 + tid]; m += v; q += v * v; }
        m *= (1.0f / B_); q = q * (1.0f / B_) - m * m;
        float r = rsqrtf(q + EPS_) * dc_g1[tid];
        sc[tid] = r; sf[tid] = dc_b1[tid] - m * r;
    }
    __syncthreads();
    for (int t = tid; t < B_ * 8; t += 256) {
        int j = t & 7;
        sh_h[t] = gelu_exact(sh_h[t] * sc[j] + sf[j]);
    }
    __syncthreads();
    if (tid < B_) {
        float s = 0.f;
        for (int j = 0; j < 8; ++j) s += dc_w2[j] * sh_h[tid * 8 + j];
        sh_s[tid] = s;
    }
    __syncthreads();
    if (tid == 0) {
        float m = 0.f, q = 0.f;
        for (int b = 0; b < B_; ++b) { float v = sh_s[b]; m += v; q += v * v; }
        m *= (1.0f / B_); q = q * (1.0f / B_) - m * m;
        float r = rsqrtf(q + EPS_) * dc_g2[0];
        sc[0] = r; sf[0] = dc_b2[0] - m * r;
    }
    __syncthreads();
    if (tid < B_) {
        float dec = sigmoidf_(sh_s[tid] * sc[0] + sf[0]);
        gate[tid] = sigmoidf_(dec - 0.5f);
    }

    // kp layer 1: [B,4] (one task per thread: 256 tasks)
    {
        int b = tid >> 2, j = tid & 3;
        float s = 0.f;
        const float* wr = kp_w1 + j * C_;
        const float* pr = p + b * C_;
        for (int c = 0; c < C_; ++c) s += wr[c] * pr[c];
        sh_a[tid] = s;
    }
    __syncthreads();
    if (tid < 4) {
        float m = 0.f, q = 0.f;
        for (int b = 0; b < B_; ++b) { float v = sh_a[b * 4 + tid]; m += v; q += v * v; }
        m *= (1.0f / B_); q = q * (1.0f / B_) - m * m;
        float r = rsqrtf(q + EPS_) * kp_g1[tid];
        sc[tid] = r; sf[tid] = kp_b1[tid] - m * r;
    }
    __syncthreads();
    {
        int j = tid & 3;
        sh_a[tid] = gelu_exact(sh_a[tid] * sc[j] + sf[j]);
    }
    __syncthreads();
    if (tid < B_ * 2) {
        int b = tid >> 1, i = tid & 1;
        float s = 0.f;
        for (int j = 0; j < 4; ++j) s += kp_w2[i * 4 + j] * sh_a[b * 4 + j];
        sh_l[tid] = s;
    }
    __syncthreads();
    if (tid < 2) {
        float m = 0.f, q = 0.f;
        for (int b = 0; b < B_; ++b) { float v = sh_l[b * 2 + tid]; m += v; q += v * v; }
        m *= (1.0f / B_); q = q * (1.0f / B_) - m * m;
        float r = rsqrtf(q + EPS_) * kp_g2[tid];
        sc[tid] = r; sf[tid] = kp_b2[tid] - m * r;
    }
    __syncthreads();
    if (tid < B_) {
        float l0 = sh_l[tid * 2 + 0] * sc[0] + sf[0];
        float l1 = sh_l[tid * 2 + 1] * sc[1] + sf[1];
        float mx = fmaxf(l0, l1);
        float e0 = expf(l0 - mx), e1 = expf(l1 - mx);
        float inv = 1.0f / (e0 + e1);
        kw0[tid] = e0 * inv;
        kw1[tid] = e1 * inv;
    }
}

// ---------------- depthwise conv helpers ----------------
__device__ __forceinline__ void dw_convs(const float* __restrict__ pl, int k,
                                         const float* w3, const float* w5,
                                         float& y3, float& y5) {
    int h = k >> 5, w = k & 31;
    float a3 = 0.f;
    #pragma unroll
    for (int dy = -1; dy <= 1; ++dy)
        #pragma unroll
        for (int dx = -1; dx <= 1; ++dx) {
            int hh = h + dy, ww = w + dx;
            if (hh >= 0 && hh < 32 && ww >= 0 && ww < 32)
                a3 += w3[(dy + 1) * 3 + (dx + 1)] * pl[hh * 32 + ww];
        }
    float a5 = 0.f;
    #pragma unroll
    for (int dy = -2; dy <= 2; ++dy)
        #pragma unroll
        for (int dx = -2; dx <= 2; ++dx) {
            int hh = h + dy, ww = w + dx;
            if (hh >= 0 && hh < 32 && ww >= 0 && ww < 32)
                a5 += w5[(dy + 2) * 5 + (dx + 2)] * pl[hh * 32 + ww];
        }
    y3 = a3; y5 = a5;
}

// pass 1: per-channel sum/sumsq of raw y3, y5 (BN training stats)
__global__ void dwstats_kernel(const float* __restrict__ x,
                               const float* __restrict__ c3w,
                               const float* __restrict__ c5w,
                               float* __restrict__ acc) {
    __shared__ float pl[HW_];
    __shared__ float a4[4];
    int bc = blockIdx.x, c = bc % C_;
    int tid = threadIdx.x;
    const float* xp = x + (size_t)bc * HW_;
    for (int k = tid; k < HW_; k += 256) pl[k] = xp[k];
    float w3[9], w5[25];
    #pragma unroll
    for (int i = 0; i < 9; ++i) w3[i] = c3w[c * 9 + i];
    #pragma unroll
    for (int i = 0; i < 25; ++i) w5[i] = c5w[c * 25 + i];
    if (tid < 4) a4[tid] = 0.f;
    __syncthreads();
    float s3 = 0.f, q3 = 0.f, s5 = 0.f, q5 = 0.f;
    for (int k = tid; k < HW_; k += 256) {
        float y3, y5;
        dw_convs(pl, k, w3, w5, y3, y5);
        s3 += y3; q3 += y3 * y3; s5 += y5; q5 += y5 * y5;
    }
    atomicAdd(&a4[0], s3); atomicAdd(&a4[1], q3);
    atomicAdd(&a4[2], s5); atomicAdd(&a4[3], q5);
    __syncthreads();
    if (tid == 0) {
        atomicAdd(&acc[0 * C_ + c], a4[0]);
        atomicAdd(&acc[1 * C_ + c], a4[1]);
        atomicAdd(&acc[2 * C_ + c], a4[2]);
        atomicAdd(&acc[3 * C_ + c], a4[3]);
    }
}

// pass 2: mix = kw0*bn(y3) + kw1*bn(y5) + (x*c1w+c1b); store, accumulate ar stats
__global__ void mix_kernel(const float* __restrict__ x,
                           const float* __restrict__ c3w, const float* __restrict__ c5w,
                           const float* c3m, const float* c3r, const float* c3g, const float* c3b,
                           const float* c5m, const float* c5r, const float* c5g, const float* c5b,
                           const float* c1w, const float* c1b,
                           const float* kw0, const float* kw1,
                           float* __restrict__ tok, float* __restrict__ acc) {
    __shared__ float pl[HW_];
    __shared__ float a2[2];
    int bc = blockIdx.x, b = bc / C_, c = bc % C_;
    int tid = threadIdx.x;
    const float* xp = x + (size_t)bc * HW_;
    for (int k = tid; k < HW_; k += 256) pl[k] = xp[k];
    float w3[9], w5[25];
    #pragma unroll
    for (int i = 0; i < 9; ++i) w3[i] = c3w[c * 9 + i];
    #pragma unroll
    for (int i = 0; i < 25; ++i) w5[i] = c5w[c * 25 + i];
    float k0 = kw0[b], k1 = kw1[b];
    float sc3 = c3r[c] * c3g[c], sf3 = c3b[c] - c3m[c] * sc3;
    float sc5 = c5r[c] * c5g[c], sf5 = c5b[c] - c5m[c] * sc5;
    float w1c = c1w[c], b1c = c1b[c];
    if (tid < 2) a2[tid] = 0.f;
    __syncthreads();
    float s = 0.f, q = 0.f;
    float* tp = tok + (size_t)bc * HW_;
    for (int k = tid; k < HW_; k += 256) {
        float y3, y5;
        dw_convs(pl, k, w3, w5, y3, y5);
        float m = k0 * (y3 * sc3 + sf3) + k1 * (y5 * sc5 + sf5) + (pl[k] * w1c + b1c);
        tp[k] = m;
        s += m; q += m * m;
    }
    atomicAdd(&a2[0], s); atomicAdd(&a2[1], q);
    __syncthreads();
    if (tid == 0) {
        atomicAdd(&acc[4 * C_ + c], a2[0]);
        atomicAdd(&acc[5 * C_ + c], a2[1]);
    }
}

// bn(tok) in place + per-(b,c) pooled mean
__global__ void normpool_kernel(float* __restrict__ tok,
                                const float* arm, const float* arr,
                                const float* ar_g, const float* ar_b,
                                float* __restrict__ gp) {
    __shared__ float red[256];
    int bc = blockIdx.x, c = bc % C_;
    float scv = arr[c] * ar_g[c];
    float sfv = ar_b[c] - arm[c] * scv;
    float* tp = tok + (size_t)bc * HW_;
    float s = 0.f;
    for (int k = threadIdx.x; k < HW_; k += 256) {
        float v = tp[k] * scv + sfv;
        tp[k] = v;
        s += v;
    }
    red[threadIdx.x] = s;
    __syncthreads();
    for (int off = 128; off; off >>= 1) {
        if (threadIdx.x < off) red[threadIdx.x] += red[threadIdx.x + off];
        __syncthreads();
    }
    if (threadIdx.x == 0) gp[bc] = red[0] * (1.0f / HW_);
}

// ---------------- SE channel gate ----------------
__global__ void cg1_kernel(const float* __restrict__ gp,
                           const float* cg_w1, const float* cg_g1, const float* cg_b1,
                           const float* cg_w2, float* __restrict__ graw) {
    __shared__ float sh[B_ * MID_];
    __shared__ float sc[MID_], sf[MID_];
    int tid = threadIdx.x;
    for (int t = tid; t < B_ * MID_; t += 256) {
        int b = t / MID_, j = t % MID_;
        float s = 0.f;
        const float* wr = cg_w1 + j * C_;
        const float* pr = gp + b * C_;
        for (int c = 0; c < C_; ++c) s += wr[c] * pr[c];
        sh[t] = s;
    }
    __syncthreads();
    if (tid < MID_) {
        float m = 0.f, q = 0.f;
        for (int b = 0; b < B_; ++b) { float v = sh[b * MID_ + tid]; m += v; q += v * v; }
        m *= (1.0f / B_); q = q * (1.0f / B_) - m * m;
        float r = rsqrtf(q + EPS_) * cg_g1[tid];
        sc[tid] = r; sf[tid] = cg_b1[tid] - m * r;
    }
    __syncthreads();
    for (int t = tid; t < B_ * MID_; t += 256) {
        int j = t % MID_;
        sh[t] = gelu_exact(sh[t] * sc[j] + sf[j]);
    }
    __syncthreads();
    for (int t = tid; t < B_ * C_; t += 256) {
        int b = t / C_, c = t % C_;
        float s = 0.f;
        const float* wr = cg_w2 + c * MID_;
        const float* hr = sh + b * MID_;
        #pragma unroll
        for (int j = 0; j < MID_; ++j) s += wr[j] * hr[j];
        graw[t] = s;
    }
}

// BN over batch per channel + sigmoid
__global__ void cg2_kernel(const float* __restrict__ graw,
                           const float* cg_g2, const float* cg_b2,
                           float* __restrict__ gse) {
    __shared__ float red[B_];
    __shared__ float mS, rS;
    int c = blockIdx.x, b = threadIdx.x;
    float v = graw[b * C_ + c];
    red[b] = v;
    __syncthreads();
    if (b == 0) {
        float m = 0.f, q = 0.f;
        for (int i = 0; i < B_; ++i) { float t = red[i]; m += t; q += t * t; }
        m *= (1.0f / B_); q = q * (1.0f / B_) - m * m;
        mS = m; rS = rsqrtf(q + EPS_);
    }
    __syncthreads();
    float n = (v - mS) * rS * cg_g2[c] + cg_b2[c];
    gse[b * C_ + c] = sigmoidf_(n);
}

// tokg -> bf16 A matrix [M, C] (NHWC transpose)
__global__ void packA_kernel(const float* __restrict__ tok,
                             const float* __restrict__ gse,
                             __bf16* __restrict__ Apack) {
    const int n = M_ * C_;
    for (int i = blockIdx.x * blockDim.x + threadIdx.x; i < n; i += gridDim.x * blockDim.x) {
        int m = i / C_, c = i % C_;
        int b = m / HW_, hw = m % HW_;
        float v = tok[((size_t)b * C_ + c) * HW_ + hw] * gse[b * C_ + c];
        Apack[i] = f2bf(v);
    }
}

// ---------------- bf16 WMMA GEMM: D[M,N] = A[M,K] * B[N,K]^T ----------------
// block: 256 threads = 8 waves; tile 128(M) x 64(N), BK=32
// wave (wm 0..3, wn 0..1) computes a 32x32 output tile = 2x2 WMMA fragments.
// Tiles are filled with CDNA5 async global->LDS copies (ASYNCcnt).
__device__ __forceinline__ v16bf load_afrag(const __bf16* rowp, int half) {
    // 16-bit A 16x32 layout: lanes 0-15 hold K 0..7 & 16..23; lanes 16-31 hold K 8..15 & 24..31
    const __bf16* rp = rowp + half * 8;
    bf16x8 x0 = *reinterpret_cast<const bf16x8*>(rp);
    bf16x8 x1 = *reinterpret_cast<const bf16x8*>(rp + 16);
    v16bf a;
    #pragma unroll
    for (int i = 0; i < 8; ++i) { a[i] = x0[i]; a[i + 8] = x1[i]; }
    return a;
}

__global__ void gemm_bf16_wmma_kernel(const __bf16* __restrict__ A,
                                      const __bf16* __restrict__ Bw,
                                      float* __restrict__ D,
                                      int M, int N, int K) {
    const int BK = 32;
    __shared__ __attribute__((aligned(16))) __bf16 sA[128 * 32];
    __shared__ __attribute__((aligned(16))) __bf16 sB[64 * 32];
    int tid  = threadIdx.x;
    int lane = tid & 31;
    int wid  = tid >> 5;
    int wm = wid & 3, wn = wid >> 2;
    int m0 = blockIdx.x * 128;
    int n0 = blockIdx.y * 64;
    int half = lane >> 4;
    int l15  = lane & 15;

    // async copy assignments: A = 1024 x 16B chunks (4/thread), B = 256 x 16B chunks (1/thread)
    unsigned sAb = (unsigned)(size_t)&sA[0];
    unsigned sBb = (unsigned)(size_t)&sB[0];
    unsigned ldsA[4];
    const __bf16* gA[4];
    #pragma unroll
    for (int j = 0; j < 4; ++j) {
        int q = tid + 256 * j;
        int row = q >> 2, col = (q & 3) * 8;
        ldsA[j] = sAb + (unsigned)(row * BK + col) * 2u;
        gA[j] = A + (size_t)(m0 + row) * K + col;
    }
    int rowB = tid >> 2, colB = (tid & 3) * 8;
    unsigned ldsB = sBb + (unsigned)(rowB * BK + colB) * 2u;
    const __bf16* gB = Bw + (size_t)(n0 + rowB) * K + colB;

    v8f acc00 = {0.f,0.f,0.f,0.f,0.f,0.f,0.f,0.f};
    v8f acc01 = {0.f,0.f,0.f,0.f,0.f,0.f,0.f,0.f};
    v8f acc10 = {0.f,0.f,0.f,0.f,0.f,0.f,0.f,0.f};
    v8f acc11 = {0.f,0.f,0.f,0.f,0.f,0.f,0.f,0.f};

    for (int kt = 0; kt < K; kt += BK) {
        __syncthreads();                      // previous fragment reads complete
        #pragma unroll
        for (int j = 0; j < 4; ++j) async_copy_b128(ldsA[j], gA[j] + kt);
        async_copy_b128(ldsB, gB + kt);
        if (kt + BK < K) {
            __builtin_prefetch(gA[0] + kt + BK, 0, 1);
            __builtin_prefetch(gB + kt + BK, 0, 1);
        }
        wait_asynccnt0();                     // ASYNCcnt == 0: LDS tile complete
        __syncthreads();

        v16bf a0 = load_afrag(&sA[(wm * 32 + l15) * BK], half);
        v16bf a1 = load_afrag(&sA[(wm * 32 + 16 + l15) * BK], half);
        // 16-bit B 32x16 layout: lane = N; lanes 0-15 hold K 0..7 & 16..23, lanes 16-31 hold K 8..15 & 24..31
        v16bf b0 = load_afrag(&sB[(wn * 32 + l15) * BK], half);
        v16bf b1 = load_afrag(&sB[(wn * 32 + 16 + l15) * BK], half);

        acc00 = __builtin_amdgcn_wmma_f32_16x16x32_bf16(false, a0, false, b0, (short)0, acc00, false, false);
        acc01 = __builtin_amdgcn_wmma_f32_16x16x32_bf16(false, a0, false, b1, (short)0, acc01, false, false);
        acc10 = __builtin_amdgcn_wmma_f32_16x16x32_bf16(false, a1, false, b0, (short)0, acc10, false, false);
        acc11 = __builtin_amdgcn_wmma_f32_16x16x32_bf16(false, a1, false, b1, (short)0, acc11, false, false);
    }

    // store: f32 C/D 16x16 layout — VGPR r: M=r (lanes 0-15), M=r+8 (lanes 16-31); N=lane&15
    int mr0 = m0 + wm * 32 + half * 8;
    int mr1 = mr0 + 16;
    int nc  = n0 + wn * 32 + l15;
    #pragma unroll
    for (int r = 0; r < 8; ++r) {
        D[(size_t)(mr0 + r) * N + nc]      = acc00[r];
        D[(size_t)(mr0 + r) * N + nc + 16] = acc01[r];
        D[(size_t)(mr1 + r) * N + nc]      = acc10[r];
        D[(size_t)(mr1 + r) * N + nc + 16] = acc11[r];
    }
}

// per-column sum/sumsq over rows (BN stats of GEMM output)
__global__ void colstats_kernel(const float* __restrict__ Z, int M, int N,
                                int rowsPerBlock, float* __restrict__ sum,
                                float* __restrict__ ss) {
    int r0 = blockIdx.x * rowsPerBlock;
    for (int col = threadIdx.x; col < N; col += blockDim.x) {
        float s = 0.f, q = 0.f;
        for (int r = 0; r < rowsPerBlock; ++r) {
            float v = Z[(size_t)(r0 + r) * N + col];
            s += v; q += v * v;
        }
        atomicAdd(&sum[col], s);
        atomicAdd(&ss[col], q);
    }
}

// h1 = bf16(gelu(bn(z1)))
__global__ void h1pack_kernel(const float* __restrict__ z1,
                              const float* mean, const float* rstd,
                              const float* g, const float* bb,
                              __bf16* __restrict__ h1b) {
    const int n = M_ * HID_;
    for (int i = blockIdx.x * blockDim.x + threadIdx.x; i < n; i += gridDim.x * blockDim.x) {
        int c = i % HID_;
        float v = (z1[i] - mean[c]) * rstd[c] * g[c] + bb[c];
        h1b[i] = f2bf(gelu_exact(v));
    }
}

// out = gate*(tok*gse + bn(z2)) + (1-gate)*x
__global__ void final_kernel(const float* __restrict__ x,
                             const float* __restrict__ tok,
                             const float* __restrict__ gse,
                             const float* __restrict__ z2,
                             const float* mean, const float* rstd,
                             const float* g2, const float* b2,
                             const float* __restrict__ gate,
                             float* __restrict__ out) {
    const int n = B_ * C_ * HW_;
    for (int i = blockIdx.x * blockDim.x + threadIdx.x; i < n; i += gridDim.x * blockDim.x) {
        int b  = i / (C_ * HW_);
        int c  = (i / HW_) % C_;
        int hw = i % HW_;
        int m  = b * HW_ + hw;
        float z = z2[(size_t)m * C_ + c];
        float zn = (z - mean[c]) * rstd[c] * g2[c] + b2[c];
        float tg = tok[i] * gse[b * C_ + c];
        float gt = gate[b];
        out[i] = gt * (tg + zn) + (1.0f - gt) * x[i];
    }
}

// ---------------- host launcher ----------------
extern "C" void kernel_launch(void* const* d_in, const int* in_sizes, int n_in,
                              void* d_out, int out_size, void* d_ws, size_t ws_size,
                              hipStream_t stream) {
    const float* x     = (const float*)d_in[0];
    const float* dc_w1 = (const float*)d_in[1];
    const float* dc_g1 = (const float*)d_in[2];
    const float* dc_b1 = (const float*)d_in[3];
    const float* dc_w2 = (const float*)d_in[4];
    const float* dc_g2 = (const float*)d_in[5];
    const float* dc_b2 = (const float*)d_in[6];
    const float* c3_w  = (const float*)d_in[7];
    const float* c3_g  = (const float*)d_in[8];
    const float* c3_b  = (const float*)d_in[9];
    const float* c5_w  = (const float*)d_in[10];
    const float* c5_g  = (const float*)d_in[11];
    const float* c5_b  = (const float*)d_in[12];
    const float* c1_w  = (const float*)d_in[13];
    const float* c1_b  = (const float*)d_in[14];
    const float* kp_w1 = (const float*)d_in[15];
    const float* kp_g1 = (const float*)d_in[16];
    const float* kp_b1 = (const float*)d_in[17];
    const float* kp_w2 = (const float*)d_in[18];
    const float* kp_g2 = (const float*)d_in[19];
    const float* kp_b2 = (const float*)d_in[20];
    const float* ar_g  = (const float*)d_in[21];
    const float* ar_b  = (const float*)d_in[22];
    const float* cg_w1 = (const float*)d_in[23];
    const float* cg_g1 = (const float*)d_in[24];
    const float* cg_b1 = (const float*)d_in[25];
    const float* cg_w2 = (const float*)d_in[26];
    const float* cg_g2 = (const float*)d_in[27];
    const float* cg_b2 = (const float*)d_in[28];
    const float* cm_w1 = (const float*)d_in[29];
    const float* cm_g1 = (const float*)d_in[30];
    const float* cm_b1 = (const float*)d_in[31];
    const float* cm_w2 = (const float*)d_in[32];
    const float* cm_g2 = (const float*)d_in[33];
    const float* cm_b2 = (const float*)d_in[34];
    float* out = (float*)d_out;

    // workspace carve (256B aligned slabs)
    char* wsp = (char*)d_ws;
    auto carve = [&](size_t bytes) -> char* {
        char* r = wsp;
        wsp += (bytes + 255) & ~(size_t)255;
        return r;
    };
    float*  p     = (float*)carve(sizeof(float) * B_ * C_);
    float*  gate  = (float*)carve(sizeof(float) * B_);
    float*  kw0   = (float*)carve(sizeof(float) * B_);
    float*  kw1   = (float*)carve(sizeof(float) * B_);
    const int ACC_LEN = 8 * C_ + 2 * HID_;                 // sums+sumsqs
    float*  acc   = (float*)carve(sizeof(float) * ACC_LEN);
    float*  fin   = (float*)carve(sizeof(float) * ACC_LEN); // means+rstds
    float*  gp    = (float*)carve(sizeof(float) * B_ * C_);
    float*  graw  = (float*)carve(sizeof(float) * B_ * C_);
    float*  gse   = (float*)carve(sizeof(float) * B_ * C_);
    float*  tok   = (float*)carve(sizeof(float) * (size_t)B_ * C_ * HW_);
    float*  z1    = (float*)carve(sizeof(float) * (size_t)M_ * HID_);
    float*  z2    = (float*)carve(sizeof(float) * (size_t)M_ * C_);
    __bf16* w1b   = (__bf16*)carve(sizeof(__bf16) * HID_ * C_);
    __bf16* w2b   = (__bf16*)carve(sizeof(__bf16) * C_ * HID_);
    __bf16* Apack = (__bf16*)carve(sizeof(__bf16) * (size_t)M_ * C_);
    __bf16* h1b   = (__bf16*)carve(sizeof(__bf16) * (size_t)M_ * HID_);

    // accumulator layout in `acc`:
    float* c3s  = acc + 0 * C_; float* c3ss = acc + 1 * C_;
    float* c5s  = acc + 2 * C_; float* c5ss = acc + 3 * C_;
    float* ars  = acc + 4 * C_; float* arss = acc + 5 * C_;
    float* cm1s = acc + 6 * C_; float* cm1ss = cm1s + HID_;
    float* cm2s = cm1ss + HID_; float* cm2ss = cm2s + C_;
    // finalized layout in `fin`:
    float* c3m = fin + 0 * C_; float* c3r = fin + 1 * C_;
    float* c5m = fin + 2 * C_; float* c5r = fin + 3 * C_;
    float* arm = fin + 4 * C_; float* arr = fin + 5 * C_;
    float* cm1m = fin + 6 * C_; float* cm1r = cm1m + HID_;
    float* cm2m = cm1r + HID_;  float* cm2r = cm2m + C_;

    const float invBHW = 1.0f / (float)M_;   // BN count over (N,H,W)

    zero_kernel<<<(ACC_LEN + 255) / 256, 256, 0, stream>>>(acc, ACC_LEN);
    pool_kernel<<<B_ * C_, 256, 0, stream>>>(x, p);
    controller_kernel<<<1, 256, 0, stream>>>(p, dc_w1, dc_g1, dc_b1, dc_w2, dc_g2, dc_b2,
                                             kp_w1, kp_g1, kp_b1, kp_w2, kp_g2, kp_b2,
                                             gate, kw0, kw1);
    dwstats_kernel<<<B_ * C_, 256, 0, stream>>>(x, c3_w, c5_w, acc);
    finalize_kernel<<<2, 256, 0, stream>>>(c3s, c3ss, invBHW, C_, c3m, c3r);
    finalize_kernel<<<2, 256, 0, stream>>>(c5s, c5ss, invBHW, C_, c5m, c5r);
    mix_kernel<<<B_ * C_, 256, 0, stream>>>(x, c3_w, c5_w,
                                            c3m, c3r, c3_g, c3_b,
                                            c5m, c5r, c5_g, c5_b,
                                            c1_w, c1_b, kw0, kw1, tok, acc);
    finalize_kernel<<<2, 256, 0, stream>>>(ars, arss, invBHW, C_, arm, arr);
    normpool_kernel<<<B_ * C_, 256, 0, stream>>>(tok, arm, arr, ar_g, ar_b, gp);
    cg1_kernel<<<1, 256, 0, stream>>>(gp, cg_w1, cg_g1, cg_b1, cg_w2, graw);
    cg2_kernel<<<C_, B_, 0, stream>>>(graw, cg_g2, cg_b2, gse);

    packA_kernel<<<4096, 256, 0, stream>>>(tok, gse, Apack);
    pack_bf16_kernel<<<512, 256, 0, stream>>>(cm_w1, w1b, HID_ * C_);
    pack_bf16_kernel<<<512, 256, 0, stream>>>(cm_w2, w2b, C_ * HID_);

    dim3 g1(M_ / 128, HID_ / 64);
    gemm_bf16_wmma_kernel<<<g1, 256, 0, stream>>>(Apack, w1b, z1, M_, HID_, C_);
    colstats_kernel<<<256, 256, 0, stream>>>(z1, M_, HID_, M_ / 256, cm1s, cm1ss);
    finalize_kernel<<<3, 256, 0, stream>>>(cm1s, cm1ss, invBHW, HID_, cm1m, cm1r);
    h1pack_kernel<<<4096, 256, 0, stream>>>(z1, cm1m, cm1r, cm_g1, cm_b1, h1b);

    dim3 g2(M_ / 128, C_ / 64);
    gemm_bf16_wmma_kernel<<<g2, 256, 0, stream>>>(h1b, w2b, z2, M_, C_, HID_);
    colstats_kernel<<<256, 256, 0, stream>>>(z2, M_, C_, M_ / 256, cm2s, cm2ss);
    finalize_kernel<<<2, 256, 0, stream>>>(cm2s, cm2ss, invBHW, C_, cm2m, cm2r);

    final_kernel<<<4096, 256, 0, stream>>>(x, tok, gse, z2, cm2m, cm2r, cm_g2, cm_b2, gate, out);
}
